// Predictor_11776800326314
// MI455X (gfx1250) — matrix operand order repaired and checked by reading, not compiled, not run on hardware
//
#include <hip/hip_runtime.h>

typedef __bf16 bf16;
typedef __attribute__((ext_vector_type(16))) __bf16        v16bf;
typedef __attribute__((ext_vector_type(8)))  float         v8f;
typedef __attribute__((ext_vector_type(4)))  unsigned int  v4u;

#define MDIM   4096
#define INROW  513
#define NPARAM 164353
#define NPAD   164360   /* padded so (b*NPAD)*2 is 16B aligned */
#define OFF_B1 131072
#define OFF_W2 131328
#define OFF_B2 164096
#define OFF_W3 164224
#define OFF_B3 164352

/* per-wave LDS scratch: H1 16x258 (padded) + H2 16x130 (padded), bf16.
   The same region doubles as the X staging buffer (16 rows x 130 f32 =
   8320 B <= 12416 B) before layer-1 compute begins. */
#define H1_STRIDE 258
#define H2_STRIDE 130
#define XS_STRIDE 130                                  /* floats per staged row */
#define WAVE_LDS  (16 * H1_STRIDE + 16 * H2_STRIDE)    /* 6208 bf16 = 12416 B */

__global__ __launch_bounds__(128) void convert_weights(
    const float* __restrict__ w, bf16* __restrict__ wb)
{
    const int b = blockIdx.y;
    const int i = (blockIdx.x * 128 + threadIdx.x) * 4;
    const float* src = w  + (size_t)b * NPARAM;
    bf16*        dst = wb + (size_t)b * NPAD;
#pragma unroll
    for (int j = 0; j < 4; ++j) {
        const int idx = i + j;
        if (idx < NPARAM) dst[idx] = (bf16)src[idx];
    }
}

__global__ __launch_bounds__(128) void mlp_fused(
    const float* __restrict__ X, const float* __restrict__ Wf,
    const bf16* __restrict__ Wb, float* __restrict__ out)
{
    __shared__ __align__(16) bf16 smem[4 * WAVE_LDS];   /* 49664 bytes */

    const int b    = blockIdx.y;
    const int m0   = blockIdx.x * 64;
    const int tid  = threadIdx.x;
    const int wave = tid >> 5;
    const int lane = tid & 31;
    const int half = lane >> 4;
    const int nco  = lane & 15;           /* A-frag row == C-frag column */

    const int row = m0 + wave * 16 + nco;
    const float* Wf_b = Wf + (size_t)b * NPARAM;
    const bf16*  Wb_b = Wb + (size_t)b * NPAD;

    bf16* h1 = smem + wave * WAVE_LDS;
    bf16* h2 = h1 + 16 * H1_STRIDE;
    float* xs = (float*)h1;               /* staging alias, dead before h1 use */

    /* LDS byte offset of this wave's staging buffer (AS3 ptrtoint) */
    const unsigned xs_base = (unsigned)(unsigned long long)
        ((__attribute__((address_space(3))) char*)(char*)(void*)xs);

    /* ---- stage X tile via async copies, 4 chunks of 128 columns; build
       layer-1 A fragments (16 rows x 512 K) from LDS ---- */
    const float* xwave = X + ((size_t)b * MDIM + m0 + wave * 16) * INROW;
    v16bf a1[16];
#pragma unroll 1
    for (int c = 0; c < 4; ++c) {
        /* coalesced async copy: one instruction = 32 lanes x 4B consecutive */
        for (int r = 0; r < 16; ++r) {
            const float* grow = xwave + (size_t)r * INROW + c * 128 + lane;
            unsigned     lrow = xs_base + (unsigned)(r * XS_STRIDE + lane) * 4u;
#pragma unroll
            for (int seg = 0; seg < 4; ++seg) {
                const unsigned long long g64 =
                    (unsigned long long)(const void*)(grow + seg * 32);
                const unsigned loff = lrow + (unsigned)(seg * 32) * 4u;
                asm volatile("global_load_async_to_lds_b32 %0, %1, off"
                             :: "v"(loff), "v"(g64) : "memory");
            }
        }
        asm volatile("s_wait_asynccnt 0x0" ::: "memory");
        /* fragment build: lane reads its own row (nco); stride 130 dwords
           puts the 16 rows on distinct LDS banks */
#pragma unroll
        for (int kt = 0; kt < 4; ++kt) {
#pragma unroll
            for (int v = 0; v < 8; ++v) {
                const int kl = kt * 32 + half * 16 + (v << 1);
                const float2 xv = *(const float2*)(xs + nco * XS_STRIDE + kl);
                a1[c * 4 + kt][2 * v]     = (bf16)xv.x;
                a1[c * 4 + kt][2 * v + 1] = (bf16)xv.y;
            }
        }
    }

    /* ---- layer 1: H1 = relu(X @ W1^T + b1), 16x256, K=512 ---- */
#pragma unroll 1
    for (int nt = 0; nt < 16; ++nt) {
        const float bias = Wf_b[OFF_B1 + nt * 16 + nco];
        v8f acc;
#pragma unroll
        for (int i = 0; i < 8; ++i) acc[i] = bias;
        const bf16* wrow = Wb_b + (size_t)(nt * 16 + nco) * 512 + half * 16;
#pragma unroll
        for (int kt = 0; kt < 16; ++kt) {
            union { v16bf h; v4u q[2]; } t;
            t.q[0] = *(const v4u*)(wrow + kt * 32);
            t.q[1] = *(const v4u*)(wrow + kt * 32 + 8);
            acc = __builtin_amdgcn_wmma_f32_16x16x32_bf16(
                      false, a1[kt], false, t.h, (short)0, acc, false, false);
        }
#pragma unroll
        for (int v = 0; v < 8; ++v) {
            float r = acc[v];
            r = r > 0.0f ? r : 0.0f;
            h1[(v + 8 * half) * H1_STRIDE + nt * 16 + nco] = (bf16)r;
        }
    }
    __syncthreads();

    /* ---- layer-2 A fragments from H1 (K = 256) ---- */
    v16bf a2[8];
#pragma unroll
    for (int kt = 0; kt < 8; ++kt) {
        union { v16bf h; unsigned u[8]; } t;
#pragma unroll
        for (int v = 0; v < 8; ++v) {
            const int k = kt * 32 + half * 16 + (v << 1);
            t.u[v] = *(const unsigned*)(h1 + nco * H1_STRIDE + k);
        }
        a2[kt] = t.h;
    }

    /* ---- layer 2: H2 = relu(H1 @ W2^T + b2), 16x128, K=256 ---- */
#pragma unroll 1
    for (int nt = 0; nt < 8; ++nt) {
        const float bias = Wf_b[OFF_B2 + nt * 16 + nco];
        v8f acc;
#pragma unroll
        for (int i = 0; i < 8; ++i) acc[i] = bias;
        const bf16* wrow = Wb_b + OFF_W2 + (size_t)(nt * 16 + nco) * 256 + half * 16;
#pragma unroll
        for (int kt = 0; kt < 8; ++kt) {
            union { v16bf h; v4u q[2]; } t;
            t.q[0] = *(const v4u*)(wrow + kt * 32);
            t.q[1] = *(const v4u*)(wrow + kt * 32 + 8);
            acc = __builtin_amdgcn_wmma_f32_16x16x32_bf16(
                      false, a2[kt], false, t.h, (short)0, acc, false, false);
        }
#pragma unroll
        for (int v = 0; v < 8; ++v) {
            float r = acc[v];
            r = r > 0.0f ? r : 0.0f;
            h2[(v + 8 * half) * H2_STRIDE + nt * 16 + nco] = (bf16)r;
        }
    }
    __syncthreads();

    /* ---- layer 3: out = sigmoid(H2 . w3 + b3), VALU dot + cross-half reduce ---- */
    float sum = 0.0f;
    const bf16*  hrow = h2 + nco * H2_STRIDE + half * 64;
    const float* w3   = Wf_b + OFF_W3 + half * 64;
#pragma unroll
    for (int j = 0; j < 64; ++j)
        sum += (float)hrow[j] * w3[j];
    sum += __shfl_xor(sum, 16, 32);
    if (lane < 16) {
        const float z = sum + Wf_b[OFF_B3];
        out[(size_t)b * MDIM + row] = 1.0f / (1.0f + __expf(-z));
    }
}

extern "C" void kernel_launch(void* const* d_in, const int* in_sizes, int n_in,
                              void* d_out, int out_size, void* d_ws, size_t ws_size,
                              hipStream_t stream)
{
    const float* X  = (const float*)d_in[0];   /* (64, 4096, 513) f32 */
    const float* Wf = (const float*)d_in[1];   /* (64, 164353)  f32 */
    float* out = (float*)d_out;                /* (64, 4096)    f32 */
    bf16*  Wb  = (bf16*)d_ws;                  /* 64 * NPAD bf16 ~ 21 MB */

    /* pass 1: weights f32 -> bf16 (L2-resident thereafter) */
    convert_weights<<<dim3((NPARAM + 511) / 512, 64), 128, 0, stream>>>(Wf, Wb);

    /* pass 2: fused 3-layer MLP, one 16-row strip per wave */
    mlp_fused<<<dim3(MDIM / 64, 64), 128, 0, stream>>>(X, Wf, Wb, out);
}
